// DepthsepCCBlock_67654324846946
// MI455X (gfx1250) — matrix operand order repaired and checked by reading, not compiled, not run on hardware
//
#include <hip/hip_runtime.h>
#include <hip/hip_bf16.h>

typedef _Float16 half_t;
typedef __attribute__((ext_vector_type(16))) _Float16 v16h;
typedef __attribute__((ext_vector_type(8)))  float    v8f;

#define NB   4
#define HH   128
#define WWID 128
#define HWL  (HH*WWID)
#define BNEPS 1e-5f

enum ConvMode {
    MODE_H16 = 0,        // write f16 NCHW (content weights, consumed by depthwise)
    MODE_H16P_RELU = 1,  // write f16 pair-packed [Co/2][HW] + relu (consumed by next conv)
    MODE_F32 = 2,        // write f32, no activation
    MODE_F32_SIG = 3,    // write f32, sigmoid
    MODE_SE_LEAKY = 4,   // f32: leaky(v * se)
    MODE_FINAL = 5       // f32: xs + leaky(v * se)
};

// pair-packed f16 layout helper: [N][C/2][HW] of half2 -> flat half index
__device__ __forceinline__ size_t pair_idx(int n, int c, int pos, int C) {
    return ((size_t)(n * (C >> 1) + (c >> 1)) * HWL + pos) * 2 + (c & 1);
}

// ---------------------------------------------------------------------------
// Implicit-GEMM conv (KS=3 SAME / KS=1 GEMM), templated; all K loops unroll.
// Xp: pair-packed f16 as u32 [NB][CIN/2][H][W]  (CIN mult of 32, padded w/ 0)
// Wp: f16 packed [Co][KS*KS][CIN]  (tap-major, channel-minor K order)
// Block = 128 threads (4 waves): tile 64 co x 64 spatial (one image row seg).
// LDS B chunk fragment-major: ldsb[col][kpair], row stride 20 dwords
// (80B, 16B aligned, conflict-free) -> B fragment = two ds_load_b128.
// ---------------------------------------------------------------------------
template<int KS, int CIN>
__launch_bounds__(128)
__global__ void wmma_conv_kernel(const uint32_t* __restrict__ Xp,
                                 const half_t* __restrict__ Wp,
                                 const float*  __restrict__ bias,
                                 half_t*   __restrict__ outh,
                                 uint32_t* __restrict__ outp,
                                 float*    __restrict__ outf,
                                 const float* __restrict__ se,
                                 const float* __restrict__ xs,
                                 int Co, int mode)
{
    constexpr int KK = KS * KS;
    __shared__ __attribute__((aligned(16))) uint32_t ldsb[64 * 20];

    const int tid  = threadIdx.x;
    const int lane = tid & 31;
    const int wv   = tid >> 5;
    const int ntile = blockIdx.x;              // 0..NB*HH*2-1
    const int w0   = (ntile & 1) << 6;
    const int hrow = (ntile >> 1) & (HH - 1);
    const int nimg = ntile >> 8;
    const int m0   = blockIdx.y * 64 + wv * 16;

    v8f acc[4];
    {
        v8f z = {0.f,0.f,0.f,0.f,0.f,0.f,0.f,0.f};
        acc[0] = z; acc[1] = z; acc[2] = z; acc[3] = z;
    }

    // A layout (ISA 16-bit A 16x32): lanes 0-15 hold K 0-7 & 16-23 of row M=lane,
    // lanes 16-31 hold K 8-15 & 24-31.
    const int arow = m0 + (lane & 15);
    const half_t* wrow = Wp + (size_t)arow * (size_t)(KK * CIN) + ((lane & 16) ? 8 : 0);
    const int colb  = lane & 15;
    const int kbase = (lane & 16) ? 4 : 0;     // B: lanes>=16 hold K 8-15 & 24-31

    #pragma unroll
    for (int j = 0; j < KK; ++j) {
        constexpr int W64 = WWID;
        const int dh = (KS == 3) ? (j / 3 - 1) : 0;
        const int dw = (KS == 3) ? (j % 3 - 1) : 0;
        const int hh = hrow + dh;
        const bool rowok = (unsigned)hh < (unsigned)HH;
        const bool interior = rowok && ((unsigned)(w0 + dw) < (unsigned)W64)
                                    && ((unsigned)(w0 + 63 + dw) < (unsigned)W64);
        const uint32_t* xrow = Xp + (size_t)nimg * (CIN / 2) * HWL
                                  + (size_t)(rowok ? hh : 0) * WWID;

        #pragma unroll
        for (int ci0 = 0; ci0 < CIN; ci0 += 32) {
            __syncthreads();
            // ---- stage B chunk (16 kpairs x 64 cols) into LDS ----
            if (interior) {
                // block-uniform: EXEC all-ones -> async direct-to-LDS loads
                #pragma unroll
                for (int i = 0; i < 8; ++i) {
                    int idx = i * 128 + tid;
                    int kp  = idx >> 6;
                    int col = idx & 63;
                    const uint32_t* gp = xrow + (size_t)(ci0 / 2 + kp) * HWL + (w0 + col + dw);
                    uint32_t ldsoff = (uint32_t)(uintptr_t)&ldsb[col * 20 + kp];
                    unsigned long long ga = (unsigned long long)(uintptr_t)gp;
                    asm volatile("global_load_async_to_lds_b32 %0, %1, off"
                                 :: "v"(ldsoff), "v"(ga) : "memory");
                }
                asm volatile("s_wait_asynccnt 0x0" ::: "memory");
            } else {
                #pragma unroll
                for (int i = 0; i < 8; ++i) {
                    int idx = i * 128 + tid;
                    int kp  = idx >> 6;
                    int col = idx & 63;
                    int wx  = w0 + col + dw;
                    uint32_t val = 0u;
                    if (rowok && (unsigned)wx < (unsigned)WWID)
                        val = xrow[(size_t)(ci0 / 2 + kp) * HWL + wx];
                    ldsb[col * 20 + kp] = val;
                }
            }
            __syncthreads();

            // ---- A fragment: two 16B loads from packed weight row ----
            union { v16h v; uint4 q[2]; } af;
            const half_t* ap = wrow + j * CIN + ci0;
            af.q[0] = *(const uint4*)(ap);
            af.q[1] = *(const uint4*)(ap + 16);
            __builtin_prefetch(ap + CIN, 0, 1);   // global_prefetch_b8: next tap

            // ---- load all 4 B fragments first (stagger LDS latency), then WMMAs
            union BF { v16h v; uint4 q[2]; } bf[4];
            #pragma unroll
            for (int nt = 0; nt < 4; ++nt) {
                const uint32_t* bp = &ldsb[(colb + nt * 16) * 20 + kbase];
                bf[nt].q[0] = *(const uint4*)(bp);
                bf[nt].q[1] = *(const uint4*)(bp + 8);
            }
            #pragma unroll
            for (int nt = 0; nt < 4; ++nt) {
                acc[nt] = __builtin_amdgcn_wmma_f32_16x16x32_f16(
                    false, af.v, false, bf[nt].v, (short)0, acc[nt], false, false);
            }
        }
    }

    // ---- epilogue: VGPR r -> M = m0 + r + (lane>=16 ? 8 : 0), N = lane&15 ----
    const int mrow = m0 + ((lane & 16) ? 8 : 0);
    #pragma unroll
    for (int nt = 0; nt < 4; ++nt) {
        int wx = w0 + nt * 16 + (lane & 15);
        int pos = hrow * WWID + wx;
        if (mode == MODE_H16P_RELU) {
            // pair-packed u32 stores: rows (mrow+2r, mrow+2r+1)
            #pragma unroll
            for (int r = 0; r < 4; ++r) {
                float v0 = acc[nt][2 * r]     + (bias ? bias[mrow + 2 * r]     : 0.f);
                float v1 = acc[nt][2 * r + 1] + (bias ? bias[mrow + 2 * r + 1] : 0.f);
                union { half_t h[2]; uint32_t u; } pk;
                pk.h[0] = (half_t)fmaxf(v0, 0.f);
                pk.h[1] = (half_t)fmaxf(v1, 0.f);
                outp[(size_t)(nimg * (Co >> 1) + (mrow >> 1) + r) * HWL + pos] = pk.u;
            }
        } else {
            #pragma unroll
            for (int r = 0; r < 8; ++r) {
                int co = mrow + r;
                float v = acc[nt][r];
                if (bias) v += bias[co];
                size_t oidx = ((size_t)nimg * Co + co) * HWL + pos;
                switch (mode) {
                    case MODE_H16:     outh[oidx] = (half_t)v; break;
                    case MODE_F32:     outf[oidx] = v; break;
                    case MODE_F32_SIG: outf[oidx] = 1.f / (1.f + __expf(-v)); break;
                    case MODE_SE_LEAKY: {
                        float t = v * se[oidx];
                        outf[oidx] = (t > 0.f) ? t : 0.2f * t;
                    } break;
                    case MODE_FINAL: {
                        float t = v * se[oidx];
                        t = (t > 0.f) ? t : 0.2f * t;
                        outf[oidx] = xs[oidx] + t;
                    } break;
                    default: break;
                }
            }
        }
    }
}

// ---------------------------------------------------------------------------
__global__ void bn_stats_kernel(const float* __restrict__ x,
                                float* __restrict__ mean, float* __restrict__ rsig, int C)
{
    __shared__ float s1[256], s2[256];
    int c = blockIdx.x, t = threadIdx.x;
    float a = 0.f, b = 0.f;
    for (int n = 0; n < NB; ++n) {
        const float* p = x + ((size_t)n * C + c) * HWL;
        for (int i = t; i < HWL; i += 256) { float v = p[i]; a += v; b += v * v; }
    }
    s1[t] = a; s2[t] = b; __syncthreads();
    for (int s = 128; s > 0; s >>= 1) {
        if (t < s) { s1[t] += s1[t + s]; s2[t] += s2[t + s]; }
        __syncthreads();
    }
    if (t == 0) {
        float cnt = (float)(NB * HWL);
        float m = s1[0] / cnt;
        float var = s2[0] / cnt - m * m;
        mean[c] = m;
        rsig[c] = rsqrtf(var + BNEPS);
    }
}

// SPADE apply + affine BN1: xsp pair-packed (conv input), xh1 NCHW (depthwise input)
__global__ void spade_apply_kernel(const float* __restrict__ x,
                                   const float* __restrict__ gam, const float* __restrict__ bet,
                                   const float* __restrict__ mean, const float* __restrict__ rsig,
                                   const float* __restrict__ g1, const float* __restrict__ b1,
                                   half_t* __restrict__ xsp, half_t* __restrict__ xh1)
{
    size_t idx = (size_t)blockIdx.x * 256 + threadIdx.x;
    size_t total = (size_t)NB * 128 * HWL;
    if (idx >= total) return;
    int pos = (int)(idx % HWL);
    int c   = (int)((idx / HWL) % 128);
    int n   = (int)(idx / ((size_t)128 * HWL));
    float xn = (x[idx] - mean[c]) * rsig[c];
    xsp[pair_idx(n, c, pos, 128)] = (half_t)(xn * (1.f + gam[idx]) + bet[idx]);
    xh1[idx] = (half_t)(xn * g1[c] + b1[c]);
}

__global__ void bn_apply_kernel(const float* __restrict__ x,
                                const float* __restrict__ mean, const float* __restrict__ rsig,
                                const float* __restrict__ g, const float* __restrict__ b,
                                half_t* __restrict__ out, int C)
{
    size_t idx = (size_t)blockIdx.x * 256 + threadIdx.x;
    size_t total = (size_t)NB * C * HWL;
    if (idx >= total) return;
    int c = (int)((idx / HWL) % C);
    out[idx] = (half_t)((x[idx] - mean[c]) * rsig[c] * g[c] + b[c]);
}

// content-adaptive depthwise 3x3; xin/cw NCHW f16, out pair-packed (conv input)
__global__ void depth_conv_kernel(const half_t* __restrict__ xin,
                                  const half_t* __restrict__ cw,
                                  half_t* __restrict__ out, int C)
{
    size_t idx = (size_t)blockIdx.x * 256 + threadIdx.x;
    size_t total = (size_t)NB * C * HWL;
    if (idx >= total) return;
    int pos = (int)(idx % HWL);
    int wx = pos % WWID, hy = pos / WWID;
    int c  = (int)((idx / HWL) % C);
    int n  = (int)(idx / ((size_t)C * HWL));
    const half_t* xc  = xin + ((size_t)n * C + c) * HWL;
    const half_t* cwp = cw + ((size_t)n * C + c) * 9 * HWL + pos;
    float s = 0.f;
    #pragma unroll
    for (int k = 0; k < 9; ++k) {
        int hh = hy + k / 3 - 1, wn = wx + k % 3 - 1;
        if ((unsigned)hh < (unsigned)HH && (unsigned)wn < (unsigned)WWID)
            s += (float)xc[hh * WWID + wn] * (float)cwp[(size_t)k * HWL];
    }
    out[pair_idx(n, c, pos, C)] = (half_t)s;
}

// power iteration for spectral sigma of 64x128 matrix, 5 iters (single block)
__global__ void sigma_kernel(const float* __restrict__ w, float* __restrict__ sig)
{
    __shared__ float su[64], sv[128], red[128], snorm;
    int t = threadIdx.x;   // 128 threads
    if (t < 64) su[t] = 0.125f;
    __syncthreads();
    for (int it = 0; it < 5; ++it) {
        float s = 0.f;
        for (int i = 0; i < 64; ++i) s += w[i * 128 + t] * su[i];
        red[t] = s * s; __syncthreads();
        for (int k = 64; k > 0; k >>= 1) { if (t < k) red[t] += red[t + k]; __syncthreads(); }
        if (t == 0) snorm = sqrtf(red[0]);
        __syncthreads();
        sv[t] = s / (snorm + 1e-12f);
        __syncthreads();
        float s2 = 0.f;
        if (t < 64) { for (int j = 0; j < 128; ++j) s2 += w[t * 128 + j] * sv[j]; }
        red[t] = (t < 64) ? s2 * s2 : 0.f; __syncthreads();
        for (int k = 64; k > 0; k >>= 1) { if (t < k) red[t] += red[t + k]; __syncthreads(); }
        if (t == 0) snorm = sqrtf(red[0]);
        __syncthreads();
        if (t < 64) su[t] = s2 / (snorm + 1e-12f);
        __syncthreads();
    }
    float s3 = 0.f;
    if (t < 64) { for (int j = 0; j < 128; ++j) s3 += w[t * 128 + j] * sv[j]; s3 *= su[t]; }
    red[t] = (t < 64) ? s3 : 0.f; __syncthreads();
    for (int k = 64; k > 0; k >>= 1) { if (t < k) red[t] += red[t + k]; __syncthreads(); }
    if (t == 0) sig[0] = red[0];
}

__global__ void cs_pack_kernel(const float* __restrict__ w, const float* __restrict__ sig,
                               half_t* __restrict__ out)
{
    int idx = blockIdx.x * 256 + threadIdx.x;
    if (idx < 64 * 128) out[idx] = (half_t)(w[idx] / sig[0]);
}

// repack conv weights OIHW f32 -> [Co][KK][CiPad] f16, zero-padded channels
__global__ void pack_w_kernel(const float* __restrict__ w, half_t* __restrict__ out,
                              int Co, int Ci, int KK, int CiPad)
{
    int idx = blockIdx.x * 256 + threadIdx.x;
    int total = Co * KK * CiPad;
    if (idx >= total) return;
    int ci = idx % CiPad;
    int j  = (idx / CiPad) % KK;
    int co = idx / (CiPad * KK);
    float v = (ci < Ci) ? w[((size_t)co * Ci + ci) * KK + j] : 0.f;
    out[idx] = (half_t)v;
}

// nearest-resize seg 256->128 + f16 + pad 35->64 channels, pair-packed u32 out
__global__ void seg_pack_kernel(const float* __restrict__ seg, uint32_t* __restrict__ out)
{
    size_t idx = (size_t)blockIdx.x * 256 + threadIdx.x;
    size_t total = (size_t)NB * 32 * HWL;   // 32 channel-pairs
    if (idx >= total) return;
    int pos = (int)(idx % HWL);
    int cp  = (int)((idx / HWL) % 32);
    int n   = (int)(idx / ((size_t)32 * HWL));
    int hy = pos / WWID, wx = pos % WWID;
    union { half_t h[2]; uint32_t u; } pk;
    int c0 = 2 * cp, c1 = 2 * cp + 1;
    pk.h[0] = (c0 < 35) ? (half_t)seg[(((size_t)n * 35 + c0) * 256 + 2 * hy) * 256 + 2 * wx] : (half_t)0.f;
    pk.h[1] = (c1 < 35) ? (half_t)seg[(((size_t)n * 35 + c1) * 256 + 2 * hy) * 256 + 2 * wx] : (half_t)0.f;
    out[idx] = pk.u;
}

// ---------------------------------------------------------------------------
extern "C" void kernel_launch(void* const* d_in, const int* in_sizes, int n_in,
                              void* d_out, int out_size, void* d_ws, size_t ws_size,
                              hipStream_t stream)
{
    const float* x      = (const float*)d_in[0];
    const float* seg    = (const float*)d_in[1];
    const float* gw1_w1 = (const float*)d_in[2];
    const float* gw1_b1 = (const float*)d_in[3];
    const float* gw1_w2 = (const float*)d_in[4];
    const float* gw1_b2 = (const float*)d_in[5];
    const float* gw2_w1 = (const float*)d_in[6];
    const float* gw2_b1 = (const float*)d_in[7];
    const float* gw2_w2 = (const float*)d_in[8];
    const float* gw2_b2 = (const float*)d_in[9];
    const float* se1_w1 = (const float*)d_in[10];
    const float* se1_b1 = (const float*)d_in[11];
    const float* se1_w2 = (const float*)d_in[12];
    const float* se1_b2 = (const float*)d_in[13];
    const float* se2_w1 = (const float*)d_in[14];
    const float* se2_b1 = (const float*)d_in[15];
    const float* se2_w2 = (const float*)d_in[16];
    const float* se2_b2 = (const float*)d_in[17];
    const float* sp_sh_w = (const float*)d_in[18];
    const float* sp_sh_b = (const float*)d_in[19];
    const float* sp_g_w  = (const float*)d_in[20];
    const float* sp_g_b  = (const float*)d_in[21];
    const float* sp_b_w  = (const float*)d_in[22];
    const float* sp_b_b  = (const float*)d_in[23];
    const float* c1_w = (const float*)d_in[24];
    const float* c1_b = (const float*)d_in[25];
    const float* c3_w = (const float*)d_in[26];
    const float* c3_b = (const float*)d_in[27];
    const float* cs_w = (const float*)d_in[28];
    const float* n1_g = (const float*)d_in[29];
    const float* n1_b = (const float*)d_in[30];
    const float* n2_g = (const float*)d_in[31];
    const float* n2_b = (const float*)d_in[32];

    char* ws = (char*)d_ws;
    size_t off = 0;
    auto alloc = [&](size_t bytes) -> void* {
        void* p = ws + off;
        off += (bytes + 255) & ~(size_t)255;
        return p;
    };
    const size_t E128 = (size_t)NB * 128 * HWL;
    const size_t E64  = (size_t)NB * 64 * HWL;

    uint32_t* segp = (uint32_t*)alloc((size_t)NB * 32 * HWL * sizeof(uint32_t));
    uint32_t* hbuf = (uint32_t*)alloc((size_t)NB * 64 * HWL * sizeof(uint32_t));   // 128 ch pair-packed
    half_t* cw1  = (half_t*)alloc((size_t)NB * 1152 * HWL * sizeof(half_t));
    half_t* cw2  = (half_t*)alloc((size_t)NB * 576 * HWL * sizeof(half_t));
    float*  se1f = (float*)alloc(E64 * sizeof(float));
    float*  se2f = (float*)alloc(E64 * sizeof(float));
    float*  gam  = (float*)alloc(E128 * sizeof(float));
    float*  bet  = (float*)alloc(E128 * sizeof(float));
    half_t* xsp  = (half_t*)alloc(E128 * sizeof(half_t));   // pair-packed
    half_t* xh1  = (half_t*)alloc(E128 * sizeof(half_t));   // NCHW
    float*  xsf  = (float*)alloc(E64 * sizeof(float));
    half_t* d1   = (half_t*)alloc(E128 * sizeof(half_t));   // pair-packed
    float*  dxa  = (float*)alloc(E64 * sizeof(float));
    half_t* xh2  = (half_t*)alloc(E64 * sizeof(half_t));    // NCHW
    half_t* d2   = (half_t*)alloc(E64 * sizeof(half_t));    // pair-packed
    float*  mean1 = (float*)alloc(128 * sizeof(float));
    float*  rs1   = (float*)alloc(128 * sizeof(float));
    float*  mean2 = (float*)alloc(64 * sizeof(float));
    float*  rs2   = (float*)alloc(64 * sizeof(float));
    float*  sig   = (float*)alloc(sizeof(float));
    half_t* pw1_gw1 = (half_t*)alloc((size_t)128 * 9 * 64 * sizeof(half_t));
    half_t* pw1_gw2 = (half_t*)alloc((size_t)128 * 9 * 64 * sizeof(half_t));
    half_t* pw1_se1 = (half_t*)alloc((size_t)128 * 9 * 64 * sizeof(half_t));
    half_t* pw1_se2 = (half_t*)alloc((size_t)128 * 9 * 64 * sizeof(half_t));
    half_t* pw1_sp  = (half_t*)alloc((size_t)128 * 9 * 64 * sizeof(half_t));
    half_t* pw2_gw1 = (half_t*)alloc((size_t)1152 * 9 * 128 * sizeof(half_t));
    half_t* pw2_gw2 = (half_t*)alloc((size_t)576 * 9 * 128 * sizeof(half_t));
    half_t* pw2_se1 = (half_t*)alloc((size_t)64 * 9 * 128 * sizeof(half_t));
    half_t* pw2_se2 = (half_t*)alloc((size_t)64 * 9 * 128 * sizeof(half_t));
    half_t* pw_g    = (half_t*)alloc((size_t)128 * 9 * 128 * sizeof(half_t));
    half_t* pw_b    = (half_t*)alloc((size_t)128 * 9 * 128 * sizeof(half_t));
    half_t* pc1     = (half_t*)alloc((size_t)64 * 128 * sizeof(half_t));
    half_t* pc3     = (half_t*)alloc((size_t)64 * 64 * sizeof(half_t));
    half_t* pcs     = (half_t*)alloc((size_t)64 * 128 * sizeof(half_t));
    (void)ws_size; (void)n_in; (void)in_sizes; (void)out_size;

    auto cdiv = [](size_t a, size_t b) { return (unsigned)((a + b - 1) / b); };

    seg_pack_kernel<<<cdiv((size_t)NB * 32 * HWL, 256), 256, 0, stream>>>(seg, segp);
    auto packw = [&](const float* w, half_t* o, int Co, int Ci, int KK, int CiPad) {
        pack_w_kernel<<<cdiv((size_t)Co * KK * CiPad, 256), 256, 0, stream>>>(w, o, Co, Ci, KK, CiPad);
    };
    packw(gw1_w1, pw1_gw1, 128, 35, 9, 64);
    packw(gw2_w1, pw1_gw2, 128, 35, 9, 64);
    packw(se1_w1, pw1_se1, 128, 35, 9, 64);
    packw(se2_w1, pw1_se2, 128, 35, 9, 64);
    packw(sp_sh_w, pw1_sp, 128, 35, 9, 64);
    packw(gw1_w2, pw2_gw1, 1152, 128, 9, 128);
    packw(gw2_w2, pw2_gw2, 576, 128, 9, 128);
    packw(se1_w2, pw2_se1, 64, 128, 9, 128);
    packw(se2_w2, pw2_se2, 64, 128, 9, 128);
    packw(sp_g_w, pw_g, 128, 128, 9, 128);
    packw(sp_b_w, pw_b, 128, 128, 9, 128);
    packw(c1_w, pc1, 64, 128, 1, 128);
    packw(c3_w, pc3, 64, 64, 1, 64);

    bn_stats_kernel<<<128, 256, 0, stream>>>(x, mean1, rs1, 128);

    const dim3 blk(128);
    auto grid = [&](int Co) { return dim3((unsigned)(NB * HH * (WWID / 64)), (unsigned)(Co / 64)); };

    // generator branches (hbuf reused sequentially on the stream)
    wmma_conv_kernel<3,64><<<grid(128), blk, 0, stream>>>(segp, pw1_gw1, gw1_b1, nullptr, hbuf, nullptr, nullptr, nullptr, 128, MODE_H16P_RELU);
    wmma_conv_kernel<3,128><<<grid(1152), blk, 0, stream>>>(hbuf, pw2_gw1, gw1_b2, cw1, nullptr, nullptr, nullptr, nullptr, 1152, MODE_H16);
    wmma_conv_kernel<3,64><<<grid(128), blk, 0, stream>>>(segp, pw1_gw2, gw2_b1, nullptr, hbuf, nullptr, nullptr, nullptr, 128, MODE_H16P_RELU);
    wmma_conv_kernel<3,128><<<grid(576), blk, 0, stream>>>(hbuf, pw2_gw2, gw2_b2, cw2, nullptr, nullptr, nullptr, nullptr, 576, MODE_H16);
    wmma_conv_kernel<3,64><<<grid(128), blk, 0, stream>>>(segp, pw1_se1, se1_b1, nullptr, hbuf, nullptr, nullptr, nullptr, 128, MODE_H16P_RELU);
    wmma_conv_kernel<3,128><<<grid(64), blk, 0, stream>>>(hbuf, pw2_se1, se1_b2, nullptr, nullptr, se1f, nullptr, nullptr, 64, MODE_F32_SIG);
    wmma_conv_kernel<3,64><<<grid(128), blk, 0, stream>>>(segp, pw1_se2, se2_b1, nullptr, hbuf, nullptr, nullptr, nullptr, 128, MODE_H16P_RELU);
    wmma_conv_kernel<3,128><<<grid(64), blk, 0, stream>>>(hbuf, pw2_se2, se2_b2, nullptr, nullptr, se2f, nullptr, nullptr, 64, MODE_F32_SIG);
    // SPADE branch: shared activation then gamma/beta
    wmma_conv_kernel<3,64><<<grid(128), blk, 0, stream>>>(segp, pw1_sp, sp_sh_b, nullptr, hbuf, nullptr, nullptr, nullptr, 128, MODE_H16P_RELU);
    wmma_conv_kernel<3,128><<<grid(128), blk, 0, stream>>>(hbuf, pw_g, sp_g_b, nullptr, nullptr, gam, nullptr, nullptr, 128, MODE_F32);
    wmma_conv_kernel<3,128><<<grid(128), blk, 0, stream>>>(hbuf, pw_b, sp_b_b, nullptr, nullptr, bet, nullptr, nullptr, 128, MODE_F32);

    spade_apply_kernel<<<cdiv(E128, 256), 256, 0, stream>>>(x, gam, bet, mean1, rs1,
                                                            n1_g, n1_b, xsp, xh1);

    sigma_kernel<<<1, 128, 0, stream>>>(cs_w, sig);
    cs_pack_kernel<<<cdiv((size_t)64 * 128, 256), 256, 0, stream>>>(cs_w, sig, pcs);
    wmma_conv_kernel<1,128><<<grid(64), blk, 0, stream>>>((const uint32_t*)xsp, pcs, nullptr, nullptr, nullptr, xsf, nullptr, nullptr, 64, MODE_F32);

    depth_conv_kernel<<<cdiv(E128, 256), 256, 0, stream>>>(xh1, cw1, d1, 128);
    wmma_conv_kernel<1,128><<<grid(64), blk, 0, stream>>>((const uint32_t*)d1, pc1, c1_b, nullptr, nullptr, dxa, se1f, nullptr, 64, MODE_SE_LEAKY);
    bn_stats_kernel<<<64, 256, 0, stream>>>(dxa, mean2, rs2, 64);
    bn_apply_kernel<<<cdiv(E64, 256), 256, 0, stream>>>(dxa, mean2, rs2, n2_g, n2_b, xh2, 64);
    depth_conv_kernel<<<cdiv(E64, 256), 256, 0, stream>>>(xh2, cw2, d2, 64);
    wmma_conv_kernel<1,64><<<grid(64), blk, 0, stream>>>((const uint32_t*)d2, pc3, c3_b, nullptr, nullptr, (float*)d_out, se2f, xsf, 64, MODE_FINAL);
}
